// GCLSTM_Model_4818953306295
// MI455X (gfx1250) — matrix operand order repaired and checked by reading, not compiled, not run on hardware
//
#include <hip/hip_runtime.h>
#include <hip/hip_bf16.h>

// ---------------------------------------------------------------------------
// GCLSTM on MI455X (gfx1250): all GEMMs via v_wmma_f32_16x16x32_bf16.
//   Gate GEMM  : [N x 512] @ [512 x 1024]  (x|h vs W_g|T_g, 4 gates fused)
//   Z GEMM     : relu(h') [N x 256] @ [256 x 256]
//   ZS/ZD GEMM : Z @ W_src, Z @ W_dst  (hoisted from edges to nodes)
//   Edge kernel: sigmoid(relu(ZS[src]+ZD[dst]) . w_out), 1 wave32 per edge.
// Tiling: block = 256 thr = 8 waves, 32 node-rows per block. Whole-K A tile
// staged once in LDS in WMMA fragment order (one barrier); each wave computes
// 2 row-tiles x 2 feature-tiles (x gates), so every B fragment is reused
// twice -> 1:1 WMMA:VMEM ratio, no barriers inside the K loop.
// ---------------------------------------------------------------------------

typedef __attribute__((ext_vector_type(16))) __bf16 v16bf;
typedef __attribute__((ext_vector_type(8)))  float  v8f;

#define WMMA_BF16(a, b, c) \
    __builtin_amdgcn_wmma_f32_16x16x32_bf16(false, (a), false, (b), (short)0, (c), false, false)

__device__ __forceinline__ unsigned short f2bf(float f) {
    union { float f; unsigned int u; } v; v.f = f;
    unsigned int u = v.u;
    u += 0x7fffu + ((u >> 16) & 1u);      // round-to-nearest-even
    return (unsigned short)(u >> 16);
}

__device__ __forceinline__ float sigm(float x) { return 1.0f / (1.0f + __expf(-x)); }

// ---------------------------------------------------------------------------
// Weight packing into WMMA B-fragment order:
//   Bpk[(((ks*G + g)*16 + ft)*32 + lane)*16 + e]  (bf16)
//   lane: N=lane&15, K-half=lane>>4;  element e -> k = ks*32 + (e>>3)*16 + (lane>>4)*8 + (e&7)
// Each lane's 16 bf16 are 32 contiguous bytes -> one 32B load per fragment.
// ---------------------------------------------------------------------------

__global__ void pack_gate_w(const float* __restrict__ W0, const float* __restrict__ W1,
                            const float* __restrict__ W2, const float* __restrict__ W3,
                            const float* __restrict__ T0, const float* __restrict__ T1,
                            const float* __restrict__ T2, const float* __restrict__ T3,
                            unsigned short* __restrict__ Bpk) {
    int t    = blockIdx.x * 256 + threadIdx.x;   // 0 .. 524287
    int e    = t & 15;
    int lane = (t >> 4) & 31;
    int ft   = (t >> 9) & 15;
    int g    = (t >> 13) & 3;
    int ks   = t >> 15;                          // 0..15
    int k    = ks * 32 + (e >> 3) * 16 + (lane >> 4) * 8 + (e & 7);   // 0..511
    int c    = ft * 16 + (lane & 15);            // 0..255
    const float* W = (g == 0) ? W0 : (g == 1) ? W1 : (g == 2) ? W2 : W3;
    const float* T = (g == 0) ? T0 : (g == 1) ? T1 : (g == 2) ? T2 : T3;
    float v = (k < 256) ? W[k * 256 + c] : T[(k - 256) * 256 + c];
    Bpk[t] = f2bf(v);
}

__global__ void pack_single_w(const float* __restrict__ W, unsigned short* __restrict__ Bpk) {
    int t    = blockIdx.x * 256 + threadIdx.x;   // 0 .. 65535
    int e    = t & 15;
    int lane = (t >> 4) & 31;
    int ft   = (t >> 9) & 15;
    int ks   = t >> 13;                          // 0..7
    int k    = ks * 32 + (e >> 3) * 16 + (lane >> 4) * 8 + (e & 7);
    int c    = ft * 16 + (lane & 15);
    Bpk[t] = f2bf(W[k * 256 + c]);
}

// Concatenate x (256) | h (256) per node, convert to bf16. [N x 512]
__global__ void pack_xh(const float* __restrict__ x, const float* __restrict__ h,
                        unsigned short* __restrict__ XH, long long total) {
    long long t = (long long)blockIdx.x * 256 + threadIdx.x;
    if (t >= total) return;
    int k = (int)(t & 511);
    long long n = t >> 9;
    float v = (k < 256) ? x[n * 256 + k] : h[n * 256 + (k - 256)];
    XH[t] = f2bf(v);
}

// ---------------------------------------------------------------------------
// A-tile staging helper (as a macro-free pattern, inlined in each kernel):
// stage 32 rows x K columns of bf16 into LDS in fragment order:
//   As[((ks*2 + m)*32 + l)*16 + hh*8 + j],  l = ((k%32)/8%2)*16 + (row%16), hh = (k%32>=16)
// Fully coalesced: piece p -> row = p>>(K/8 bits), chunk c = p % (K/8),
// source = A[row*K + c*8 .. +8)  (16B aligned).
// ---------------------------------------------------------------------------

// Fused gate GEMM + LSTM pointwise epilogue. K=512, 4 gates.
__global__ void __launch_bounds__(256) gates_gemm(
    const unsigned short* __restrict__ XH, const unsigned short* __restrict__ Bpk,
    const float* __restrict__ c0,
    const float* __restrict__ bconv_i, const float* __restrict__ b_i,
    const float* __restrict__ bconv_f, const float* __restrict__ b_f,
    const float* __restrict__ bconv_c, const float* __restrict__ b_c,
    const float* __restrict__ bconv_o, const float* __restrict__ b_o,
    const float* __restrict__ wc_i, const float* __restrict__ wc_f,
    const float* __restrict__ wc_o,
    float* __restrict__ out_h, float* __restrict__ out_c,
    unsigned short* __restrict__ Hrelu, int N) {
    __shared__ alignas(32) unsigned short As[16384];   // 32 rows x 512 K bf16, frag order
    const int tid = threadIdx.x, wave = tid >> 5, lane = tid & 31;
    const int rt = blockIdx.x * 32;

    // Stage whole A tile once (2048 x 16B pieces, 8 per thread, coalesced).
#pragma unroll
    for (int i = 0; i < 8; ++i) {
        const int p = i * 256 + tid;
        const int r = p >> 6, c = p & 63;            // row 0..31, 8-elem chunk 0..63
        int row = rt + r; if (row >= N) row = N - 1;
        const uint4 v = *(const uint4*)(XH + (long long)row * 512 + c * 8);
        const int ks = c >> 2, rem = c & 3;
        const int l  = (rem & 1) * 16 + (r & 15);
        *(uint4*)(&As[((ks * 2 + (r >> 4)) * 32 + l) * 16 + (rem >> 1) * 8]) = v;
    }
    __syncthreads();

    v8f acc[2][2][4] = {};                            // [mtile][ftile][gate]
    for (int ks = 0; ks < 16; ++ks) {
        const v16bf a0 = *(const v16bf*)(&As[((ks * 2 + 0) * 32 + lane) * 16]);
        const v16bf a1 = *(const v16bf*)(&As[((ks * 2 + 1) * 32 + lane) * 16]);
#pragma unroll
        for (int fp = 0; fp < 2; ++fp) {
            const int ft = wave * 2 + fp;
#pragma unroll
            for (int g = 0; g < 4; ++g) {
                const v16bf b = *(const v16bf*)(Bpk + (((ks * 4 + g) * 16 + ft) * 32 + lane) * 16);
                acc[0][fp][g] = WMMA_BF16(a0, b, acc[0][fp][g]);
                acc[1][fp][g] = WMMA_BF16(a1, b, acc[1][fp][g]);
            }
        }
    }

    // Epilogue: lane owns column col; VGPR v owns row v + 8*(lane>>4) of m-tile.
    const int hi = lane >> 4, nn = lane & 15;
#pragma unroll
    for (int fp = 0; fp < 2; ++fp) {
        const int col = (wave * 2 + fp) * 16 + nn;
        const float si = bconv_i[col] + b_i[col];
        const float sf = bconv_f[col] + b_f[col];
        const float sc = bconv_c[col] + b_c[col];
        const float so = bconv_o[col] + b_o[col];
        const float wi = wc_i[col], wf = wc_f[col], wo = wc_o[col];
#pragma unroll
        for (int m = 0; m < 2; ++m) {
#pragma unroll
            for (int v = 0; v < 8; ++v) {
                const int node = rt + m * 16 + v + 8 * hi;
                if (node < N) {
                    const long long idx = (long long)node * 256 + col;
                    const float cp = c0[idx];
                    const float ig = sigm(acc[m][fp][0][v] + si + wi * cp);
                    const float fg = sigm(acc[m][fp][1][v] + sf + wf * cp);
                    const float tg = tanhf(acc[m][fp][2][v] + sc);
                    const float cn = fg * cp + ig * tg;
                    const float og = sigm(acc[m][fp][3][v] + so + wo * cn);
                    const float hn = og * tanhf(cn);
                    out_h[idx] = hn;
                    out_c[idx] = cn;
                    Hrelu[idx] = f2bf(hn > 0.f ? hn : 0.f);
                }
            }
        }
    }
}

// Z = relu(h') @ W_lin + b_lin -> bf16.  K=256.
__global__ void __launch_bounds__(256) z_gemm(
    const unsigned short* __restrict__ Abf, const unsigned short* __restrict__ Bpk,
    const float* __restrict__ b_lin, unsigned short* __restrict__ Zbf, int N) {
    __shared__ alignas(32) unsigned short As[8192];    // 32 rows x 256 K
    const int tid = threadIdx.x, wave = tid >> 5, lane = tid & 31;
    const int rt = blockIdx.x * 32;
#pragma unroll
    for (int i = 0; i < 4; ++i) {
        const int p = i * 256 + tid;
        const int r = p >> 5, c = p & 31;
        int row = rt + r; if (row >= N) row = N - 1;
        const uint4 v = *(const uint4*)(Abf + (long long)row * 256 + c * 8);
        const int ks = c >> 2, rem = c & 3;
        const int l  = (rem & 1) * 16 + (r & 15);
        *(uint4*)(&As[((ks * 2 + (r >> 4)) * 32 + l) * 16 + (rem >> 1) * 8]) = v;
    }
    __syncthreads();

    v8f acc[2][2] = {};
    for (int ks = 0; ks < 8; ++ks) {
        const v16bf a0 = *(const v16bf*)(&As[((ks * 2 + 0) * 32 + lane) * 16]);
        const v16bf a1 = *(const v16bf*)(&As[((ks * 2 + 1) * 32 + lane) * 16]);
#pragma unroll
        for (int fp = 0; fp < 2; ++fp) {
            const int ft = wave * 2 + fp;
            const v16bf b = *(const v16bf*)(Bpk + ((ks * 16 + ft) * 32 + lane) * 16);
            acc[0][fp] = WMMA_BF16(a0, b, acc[0][fp]);
            acc[1][fp] = WMMA_BF16(a1, b, acc[1][fp]);
        }
    }
    const int hi = lane >> 4, nn = lane & 15;
#pragma unroll
    for (int fp = 0; fp < 2; ++fp) {
        const int col = (wave * 2 + fp) * 16 + nn;
        const float bb = b_lin[col];
#pragma unroll
        for (int m = 0; m < 2; ++m) {
#pragma unroll
            for (int v = 0; v < 8; ++v) {
                const int node = rt + m * 16 + v + 8 * hi;
                if (node < N) Zbf[(long long)node * 256 + col] = f2bf(acc[m][fp][v] + bb);
            }
        }
    }
}

// ZS = Z@W_src + b_src,  ZD = Z@W_dst + b_dst  (f32 outputs). K=256.
__global__ void __launch_bounds__(256) zsd_gemm(
    const unsigned short* __restrict__ Zbf,
    const unsigned short* __restrict__ BpkS, const unsigned short* __restrict__ BpkD,
    const float* __restrict__ b_src, const float* __restrict__ b_dst,
    float* __restrict__ ZS, float* __restrict__ ZD, int N) {
    __shared__ alignas(32) unsigned short As[8192];
    const int tid = threadIdx.x, wave = tid >> 5, lane = tid & 31;
    const int rt = blockIdx.x * 32;
#pragma unroll
    for (int i = 0; i < 4; ++i) {
        const int p = i * 256 + tid;
        const int r = p >> 5, c = p & 31;
        int row = rt + r; if (row >= N) row = N - 1;
        const uint4 v = *(const uint4*)(Zbf + (long long)row * 256 + c * 8);
        const int ks = c >> 2, rem = c & 3;
        const int l  = (rem & 1) * 16 + (r & 15);
        *(uint4*)(&As[((ks * 2 + (r >> 4)) * 32 + l) * 16 + (rem >> 1) * 8]) = v;
    }
    __syncthreads();

    v8f accS[2][2] = {}, accD[2][2] = {};
    for (int ks = 0; ks < 8; ++ks) {
        const v16bf a0 = *(const v16bf*)(&As[((ks * 2 + 0) * 32 + lane) * 16]);
        const v16bf a1 = *(const v16bf*)(&As[((ks * 2 + 1) * 32 + lane) * 16]);
#pragma unroll
        for (int fp = 0; fp < 2; ++fp) {
            const int ft = wave * 2 + fp;
            const v16bf bs = *(const v16bf*)(BpkS + ((ks * 16 + ft) * 32 + lane) * 16);
            const v16bf bd = *(const v16bf*)(BpkD + ((ks * 16 + ft) * 32 + lane) * 16);
            accS[0][fp] = WMMA_BF16(a0, bs, accS[0][fp]);
            accS[1][fp] = WMMA_BF16(a1, bs, accS[1][fp]);
            accD[0][fp] = WMMA_BF16(a0, bd, accD[0][fp]);
            accD[1][fp] = WMMA_BF16(a1, bd, accD[1][fp]);
        }
    }
    const int hi = lane >> 4, nn = lane & 15;
#pragma unroll
    for (int fp = 0; fp < 2; ++fp) {
        const int col = (wave * 2 + fp) * 16 + nn;
        const float bs = b_src[col], bd = b_dst[col];
#pragma unroll
        for (int m = 0; m < 2; ++m) {
#pragma unroll
            for (int v = 0; v < 8; ++v) {
                const int node = rt + m * 16 + v + 8 * hi;
                if (node < N) {
                    const long long idx = (long long)node * 256 + col;
                    ZS[idx] = accS[m][fp][v] + bs;
                    ZD[idx] = accD[m][fp][v] + bd;
                }
            }
        }
    }
}

// One wave32 per edge: gather ZS[src], ZD[dst], ZD[neg] rows (8 floats/lane),
// relu+dot with w_out, wave shuffle reduction, sigmoid.
__global__ void __launch_bounds__(256) edge_pred(
    const int* __restrict__ src, const int* __restrict__ dst, const int* __restrict__ neg,
    const float* __restrict__ ZS, const float* __restrict__ ZD,
    const float* __restrict__ W_out, const float* __restrict__ b_out,
    float* __restrict__ pos_out, float* __restrict__ neg_out, int E) {
    const int lane = threadIdx.x & 31;
    const int e = blockIdx.x * 8 + (threadIdx.x >> 5);
    if (e >= E) return;
    const int s = src[e], d = dst[e], n = neg[e];
    const float4* ps = (const float4*)(ZS + (long long)s * 256 + lane * 8);
    const float4* pd = (const float4*)(ZD + (long long)d * 256 + lane * 8);
    const float4* pn = (const float4*)(ZD + (long long)n * 256 + lane * 8);
    const float4* pw = (const float4*)(W_out + lane * 8);
    float accP = 0.f, accN = 0.f;
#pragma unroll
    for (int q = 0; q < 2; ++q) {
        const float4 a = ps[q], bq = pd[q], cq = pn[q], w = pw[q];
        accP += fmaxf(a.x + bq.x, 0.f) * w.x + fmaxf(a.y + bq.y, 0.f) * w.y +
                fmaxf(a.z + bq.z, 0.f) * w.z + fmaxf(a.w + bq.w, 0.f) * w.w;
        accN += fmaxf(a.x + cq.x, 0.f) * w.x + fmaxf(a.y + cq.y, 0.f) * w.y +
                fmaxf(a.z + cq.z, 0.f) * w.z + fmaxf(a.w + cq.w, 0.f) * w.w;
    }
#pragma unroll
    for (int off = 16; off > 0; off >>= 1) {
        accP += __shfl_down(accP, off, 32);
        accN += __shfl_down(accN, off, 32);
    }
    if (lane == 0) {
        const float b0 = b_out[0];
        pos_out[e] = sigm(accP + b0);
        neg_out[e] = sigm(accN + b0);
    }
}

// ---------------------------------------------------------------------------
// Host-side orchestration.
// d_in order: node_feat, edge_weight, h_0, c_0, src, dst, neg, then params in
// dict-insertion order: (W_g,T_g,bconv_g,b_g) for g in i,f,c,o; wc_i,wc_f,wc_o;
// W_lin,b_lin,W_src,b_src,W_dst,b_dst,W_out,b_out.
// d_out: [E pos][E neg][N*256 h_new][N*256 c_new]
// ---------------------------------------------------------------------------
extern "C" void kernel_launch(void* const* d_in, const int* in_sizes, int n_in,
                              void* d_out, int out_size, void* d_ws, size_t ws_size,
                              hipStream_t stream) {
    const float* x  = (const float*)d_in[0];
    const float* h0 = (const float*)d_in[2];
    const float* c0 = (const float*)d_in[3];
    const int* src  = (const int*)d_in[4];
    const int* dst  = (const int*)d_in[5];
    const int* neg  = (const int*)d_in[6];

    const float *W[4], *T[4], *bconv[4], *bg[4];
    for (int g = 0; g < 4; ++g) {
        W[g]     = (const float*)d_in[7 + 4 * g];
        T[g]     = (const float*)d_in[8 + 4 * g];
        bconv[g] = (const float*)d_in[9 + 4 * g];
        bg[g]    = (const float*)d_in[10 + 4 * g];
    }
    const float* wc_i  = (const float*)d_in[23];
    const float* wc_f  = (const float*)d_in[24];
    const float* wc_o  = (const float*)d_in[25];
    const float* W_lin = (const float*)d_in[26];
    const float* b_lin = (const float*)d_in[27];
    const float* W_src = (const float*)d_in[28];
    const float* b_src = (const float*)d_in[29];
    const float* W_dst = (const float*)d_in[30];
    const float* b_dst = (const float*)d_in[31];
    const float* W_out = (const float*)d_in[32];
    const float* b_out = (const float*)d_in[33];

    const int N = in_sizes[0] / 256;
    const int E = in_sizes[4];

    // Workspace layout (ZS reuses the XH region once gates are done).
    char* ws = (char*)d_ws;
    unsigned short* BpkG = (unsigned short*)(ws);                     // 1 MB
    unsigned short* BpkL = (unsigned short*)(ws + (1u << 20));        // 128 KB
    unsigned short* BpkS = (unsigned short*)(ws + (1u << 20) + 131072);
    unsigned short* BpkD = (unsigned short*)(ws + (1u << 20) + 2 * 131072);
    size_t off = 2u << 20;
    unsigned short* XH = (unsigned short*)(ws + off);                 // N*512 bf16
    float* ZS          = (float*)(ws + off);                          // N*256 f32 (reuse)
    size_t szXH = (size_t)N * 512 * 2, szZS = (size_t)N * 256 * 4;
    size_t offH = off + (szXH > szZS ? szXH : szZS);
    unsigned short* Hrelu = (unsigned short*)(ws + offH);             // N*256 bf16
    unsigned short* Zbf   = (unsigned short*)(ws + offH + (size_t)N * 256 * 2);
    float* ZD             = (float*)(ws + offH + (size_t)N * 256 * 2 + (size_t)N * 256 * 2);

    float* out   = (float*)d_out;
    float* pos_o = out;
    float* neg_o = out + E;
    float* out_h = out + 2 * (size_t)E;
    float* out_c = out + 2 * (size_t)E + (size_t)N * 256;

    // 1) Pack weights into WMMA B-fragment order (bf16).
    pack_gate_w<<<2048, 256, 0, stream>>>(W[0], W[1], W[2], W[3], T[0], T[1], T[2], T[3], BpkG);
    pack_single_w<<<256, 256, 0, stream>>>(W_lin, BpkL);
    pack_single_w<<<256, 256, 0, stream>>>(W_src, BpkS);
    pack_single_w<<<256, 256, 0, stream>>>(W_dst, BpkD);

    // 2) Pack [x|h] -> bf16.
    const long long totXH = (long long)N * 512;
    pack_xh<<<(unsigned)((totXH + 255) / 256), 256, 0, stream>>>(x, h0, XH, totXH);

    // 3) Fused gate GEMM + LSTM epilogue (32 rows / block).
    const int rtiles = (N + 31) / 32;
    gates_gemm<<<rtiles, 256, 0, stream>>>(XH, BpkG, c0,
                                           bconv[0], bg[0], bconv[1], bg[1],
                                           bconv[2], bg[2], bconv[3], bg[3],
                                           wc_i, wc_f, wc_o,
                                           out_h, out_c, Hrelu, N);

    // 4) Z = relu(h')@W_lin + b_lin (bf16 for next GEMM).
    z_gemm<<<rtiles, 256, 0, stream>>>(Hrelu, BpkL, b_lin, Zbf, N);

    // 5) ZS / ZD per node (hoisted link-pred GEMMs).
    zsd_gemm<<<rtiles, 256, 0, stream>>>(Zbf, BpkS, BpkD, b_src, b_dst, ZS, ZD, N);

    // 6) Per-edge gather + dot + sigmoid.
    edge_pred<<<(E + 7) / 8, 256, 0, stream>>>(src, dst, neg, ZS, ZD, W_out, b_out,
                                               pos_o, neg_o, E);
}
